// EstCoordNet_3066606649362
// MI455X (gfx1250) — compile-verified, matrix operand order checked by reading, not run on hardware
//
#include <hip/hip_runtime.h>

#define BB 8
#define NN 2048
#define MT (BB * NN) /* 16384 points */

typedef __attribute__((ext_vector_type(16))) _Float16 v16h;
typedef __attribute__((ext_vector_type(8)))  float    v8f;

union Frag16 { v16h h; unsigned int u[8]; };

// ---------------- utility / packing kernels ----------------

__global__ __launch_bounds__(256) void k_zerof(float* p, int n) {
  int i = blockIdx.x * 256 + threadIdx.x;
  if (i < n) p[i] = 0.f;
}

__global__ __launch_bounds__(256) void k_pad_w(const float* __restrict__ W, _Float16* __restrict__ Wh,
                                               int Cout, int Cin, int CoutP, int CinP) {
  int idx = blockIdx.x * 256 + threadIdx.x;
  if (idx >= CoutP * CinP) return;
  int co = idx / CinP, ci = idx - co * CinP;
  float v = (co < Cout && ci < Cin) ? W[co * Cin + ci] : 0.f;
  Wh[idx] = (_Float16)v;
}

__global__ __launch_bounds__(256) void k_pad_b(const float* __restrict__ b, float* __restrict__ bp, int C, int CP) {
  int i = blockIdx.x * 256 + threadIdx.x;
  if (i < CP) bp[i] = (i < C) ? b[i] : 0.f;
}

__global__ __launch_bounds__(256) void k_x0(const float* __restrict__ pc, _Float16* __restrict__ x0) {
  int idx = blockIdx.x * 256 + threadIdx.x;
  if (idx >= MT * 32) return;
  int i = idx >> 5, c = idx & 31;
  x0[idx] = (_Float16)((c < 3) ? pc[i * 3 + c] : 0.f);
}

// ---------------- WMMA GEMM ----------------
// Y[M,Npad] = A[M,K](f16) * W[Npad,K](f16)^T + bias, fp32 accumulate.
// One wave owns a 32 x (16*NSUB) output tile: A fragments (2 M-subtiles) are
// reused across NSUB B fragments -> 2*NSUB v_wmma_f32_16x16x32_f16 per k-step.

template <int NSUB>
__global__ __launch_bounds__(256) void k_gemm(const _Float16* __restrict__ A,
                                              const _Float16* __restrict__ W,
                                              const float* __restrict__ bias,
                                              float* __restrict__ Y,
                                              int M, int Npad, int K) {
  const int TN = 16 * NSUB;
  int wave = blockIdx.x * 8 + (threadIdx.x >> 5);
  int tilesN = Npad / TN;
  int tileM = wave / tilesN;
  int tileN = wave - tileM * tilesN;
  if (tileM * 32 >= M) return;                  // whole-wave uniform exit; EXEC stays full
  int lane = threadIdx.x & 31;
  int hl = lane >> 4;                           // lane half
  int r  = lane & 15;
  const _Float16* arow0 = A + (size_t)(tileM * 32 + r) * K;
  const _Float16* arow1 = arow0 + (size_t)16 * K;
  const _Float16* wr[NSUB];
#pragma unroll
  for (int s = 0; s < NSUB; ++s)
    wr[s] = W + (size_t)(tileN * TN + s * 16 + r) * K;

  v8f acc0[NSUB], acc1[NSUB];
#pragma unroll
  for (int s = 0; s < NSUB; ++s) { acc0[s] = (v8f){}; acc1[s] = (v8f){}; }

  for (int k0 = 0; k0 < K; k0 += 32) {
    Frag16 a0, a1;
    const unsigned int* ap0 = (const unsigned int*)(arow0 + k0);
    const unsigned int* ap1 = (const unsigned int*)(arow1 + k0);
#pragma unroll
    for (int j = 0; j < 8; ++j) {
      // 16-bit A layout: VGPR j holds K pair at 16*(j>>2) + 8*hl + 2*(j&3) -> dword:
      int dw = ((j >> 2) << 3) + (hl << 2) + (j & 3);
      a0.u[j] = ap0[dw];
      a1.u[j] = ap1[dw];
    }
    if (k0 + 64 < K) {                           // gfx1250 global_prefetch_b8
      __builtin_prefetch((const void*)(arow0 + k0 + 64), 0, 3);
      __builtin_prefetch((const void*)(arow1 + k0 + 64), 0, 3);
    }
#pragma unroll
    for (int s = 0; s < NSUB; ++s) {
      Frag16 bf;
      const unsigned int* wp = (const unsigned int*)(wr[s] + k0);
#pragma unroll
      for (int j = 0; j < 8; ++j) {
        int dw = ((j >> 2) << 3) + (hl << 2) + (j & 3);
        bf.u[j] = wp[dw];
      }
      acc0[s] = __builtin_amdgcn_wmma_f32_16x16x32_f16(false, a0.h, false, bf.h,
                                                       (short)0, acc0[s], false, false);
      acc1[s] = __builtin_amdgcn_wmma_f32_16x16x32_f16(false, a1.h, false, bf.h,
                                                       (short)0, acc1[s], false, false);
    }
  }
#pragma unroll
  for (int s = 0; s < NSUB; ++s) {
    int col = tileN * TN + s * 16 + r;
    float bv = bias[col];
#pragma unroll
    for (int v = 0; v < 8; ++v) {
      int row0 = tileM * 32 + (hl << 3) + v;     // C/D layout: col = r, row = v + 8*hl
      Y[(size_t)row0 * Npad + col]        = acc0[s][v] + bv;
      Y[(size_t)(row0 + 16) * Npad + col] = acc1[s][v] + bv;
    }
  }
}

// ---------------- batch-norm (training stats), coalesced ----------------
// pass 1: block = 16 consecutive channels x one 1024-row stripe -> partial sum/sumsq
// lanes 0..15 read 16 adjacent channels of one row (64B contiguous).

__global__ __launch_bounds__(256) void k_bnpart(const float* __restrict__ Y,
                                                float* __restrict__ part, int C) {
  int cBase = blockIdx.x << 4;
  int rs = blockIdx.y;                 // 16 row chunks of MT/16 = 1024 rows
  int c16 = threadIdx.x & 15;
  int rg  = threadIdx.x >> 4;          // 0..15
  int c = cBase + c16;
  float s = 0.f, s2 = 0.f;
  int i0 = rs << 10;
  for (int i = i0 + rg; i < i0 + 1024; i += 16) {
    float v = Y[(size_t)i * C + c];
    s += v; s2 += v * v;
  }
  __shared__ float sh[256], sh2[256];
  sh[threadIdx.x] = s; sh2[threadIdx.x] = s2;
  __syncthreads();
  for (int off = 8; off > 0; off >>= 1) {
    if (rg < off) {
      sh[threadIdx.x]  += sh[threadIdx.x + (off << 4)];
      sh2[threadIdx.x] += sh2[threadIdx.x + (off << 4)];
    }
    __syncthreads();
  }
  if (rg == 0) {
    part[rs * C + c]        = sh[threadIdx.x];
    part[(16 + rs) * C + c] = sh2[threadIdx.x];
  }
}

__global__ __launch_bounds__(256) void k_bnfin(const float* __restrict__ part,
    const float* __restrict__ gamma, const float* __restrict__ beta,
    float* __restrict__ scale, float* __restrict__ shift, int C) {
  int c = blockIdx.x * 256 + threadIdx.x;
  if (c >= C) return;
  float s = 0.f, s2 = 0.f;
  for (int rs = 0; rs < 16; ++rs) { s += part[rs * C + c]; s2 += part[(16 + rs) * C + c]; }
  float mean = s * (1.f / MT);
  float var  = fmaxf(s2 * (1.f / MT) - mean * mean, 0.f);
  float sc = gamma[c] * rsqrtf(var + 1e-5f);
  scale[c] = sc;
  shift[c] = beta[c] - mean * sc;
}

__global__ __launch_bounds__(256) void k_bnrelu(const float* __restrict__ Y,
    const float* __restrict__ scale, const float* __restrict__ shift,
    _Float16* __restrict__ dA, int ldA, _Float16* __restrict__ dB, int ldB, int colOff, int C) {
  int idx = blockIdx.x * 256 + threadIdx.x;
  if (idx >= MT * C) return;
  int i = idx / C, c = idx - i * C;
  float v = fmaxf(Y[idx] * scale[c] + shift[c], 0.f);
  _Float16 h = (_Float16)v;
  dA[(size_t)i * ldA + c] = h;
  if (dB) dB[(size_t)i * ldB + colOff + c] = h;
}

// max over N of relu(bn(y)) per (batch, channel), coalesced like k_bnpart
__global__ __launch_bounds__(256) void k_maxpool(const float* __restrict__ Y,
    const float* __restrict__ scale, const float* __restrict__ shift, float* __restrict__ g) {
  int b = blockIdx.y;
  int cBase = blockIdx.x << 4;
  int c16 = threadIdx.x & 15;
  int ng  = threadIdx.x >> 4;
  int c = cBase + c16;
  float sc = scale[c], sf = shift[c];
  float m = -3.4e38f;
  const float* Yb = Y + (size_t)b * NN * 1024;
  for (int n = ng; n < NN; n += 16)
    m = fmaxf(m, Yb[(size_t)n * 1024 + c] * sc + sf);
  __shared__ float sh[256];
  sh[threadIdx.x] = m; __syncthreads();
  for (int off = 8; off > 0; off >>= 1) {
    if (ng < off) sh[threadIdx.x] = fmaxf(sh[threadIdx.x], sh[threadIdx.x + (off << 4)]);
    __syncthreads();
  }
  if (ng == 0) g[b * 1024 + c] = fmaxf(sh[threadIdx.x], 0.f);  // relu(max) == max(relu)
}

__global__ __launch_bounds__(256) void k_bcast(const float* __restrict__ g, _Float16* __restrict__ Xc) {
  int idx = blockIdx.x * 256 + threadIdx.x;
  if (idx >= MT * 1024) return;
  int i = idx >> 10, c = idx & 1023;
  int b = i >> 11;
  Xc[(size_t)i * 1088 + 64 + c] = (_Float16)g[b * 1024 + c];
}

// ---------------- losses ----------------

__global__ __launch_bounds__(256) void k_minrows(const float* __restrict__ P, int ldP,
    const float* __restrict__ Q, int ldQ, float* __restrict__ outMin) {
  int b = blockIdx.x >> 11, i = blockIdx.x & 2047;
  const float* p = P + ((size_t)b * NN + i) * ldP;
  float px = p[0], py = p[1], pz = p[2];
  const float* qb = Q + (size_t)b * NN * ldQ;
  float best = 3.4e38f;
  for (int j = threadIdx.x; j < NN; j += 256) {
    const float* q = qb + (size_t)j * ldQ;
    float dx = px - q[0], dy = py - q[1], dz = pz - q[2];
    best = fminf(best, dx * dx + dy * dy + dz * dz);
  }
  __shared__ float sh[256];
  sh[threadIdx.x] = best; __syncthreads();
  for (int off = 128; off > 0; off >>= 1) {
    if (threadIdx.x < off) sh[threadIdx.x] = fminf(sh[threadIdx.x], sh[threadIdx.x + off]);
    __syncthreads();
  }
  if (threadIdx.x == 0) outMin[blockIdx.x] = sqrtf(fmaxf(sh[0], 0.f) + 1e-24f);
}

__global__ __launch_bounds__(256) void k_summax(const float* __restrict__ m12,
    const float* __restrict__ m21, float* __restrict__ accSum, float* __restrict__ maxB) {
  int arr = blockIdx.x >> 3, b = blockIdx.x & 7;
  const float* v = (arr ? m21 : m12) + b * NN;
  float s = 0.f, mx = 0.f;
  for (int i = threadIdx.x; i < NN; i += 256) { float x = v[i]; s += x; mx = fmaxf(mx, x); }
  __shared__ float sh[256], shm[256];
  sh[threadIdx.x] = s; shm[threadIdx.x] = mx; __syncthreads();
  for (int off = 128; off > 0; off >>= 1) {
    if (threadIdx.x < off) {
      sh[threadIdx.x] += sh[threadIdx.x + off];
      shm[threadIdx.x] = fmaxf(shm[threadIdx.x], shm[threadIdx.x + off]);
    }
    __syncthreads();
  }
  if (threadIdx.x == 0) { atomicAdd(&accSum[arr], sh[0]); maxB[blockIdx.x] = shm[0]; }
}

// repulsion: 6 nearest (incl. self), drop smallest, sum max(0, 0.01 - sqrt(d2+eps))
__global__ __launch_bounds__(256) void k_rep(const float* __restrict__ pred, float* __restrict__ accRep) {
  int idx = blockIdx.x * 256 + threadIdx.x;       // exactly MT threads
  int b = idx >> 11, i = idx & 2047;
  const float* base = pred + (size_t)b * NN * 16;
  const float* p = base + (size_t)i * 16;
  float px = p[0], py = p[1], pz = p[2];
  float top[6];
#pragma unroll
  for (int t = 0; t < 6; ++t) top[t] = 3.4e38f;
  for (int j = 0; j < NN; ++j) {
    const float* q = base + (size_t)j * 16;
    float dx = px - q[0], dy = py - q[1], dz = pz - q[2];
    float d2 = dx * dx + dy * dy + dz * dz;
    if (d2 < top[5]) {
      top[5] = d2;
#pragma unroll
      for (int t = 5; t > 0; --t)
        if (top[t] < top[t - 1]) { float tmp = top[t]; top[t] = top[t - 1]; top[t - 1] = tmp; }
    }
  }
  float s = 0.f;
#pragma unroll
  for (int t = 1; t < 6; ++t) {
    float d = sqrtf(fmaxf(top[t], 0.f) + 1e-12f);
    s += fmaxf(0.01f - d, 0.f);
  }
  __shared__ float sh[256];
  sh[threadIdx.x] = s; __syncthreads();
  for (int off = 128; off > 0; off >>= 1) {
    if (threadIdx.x < off) sh[threadIdx.x] += sh[threadIdx.x + off];
    __syncthreads();
  }
  if (threadIdx.x == 0) atomicAdd(accRep, sh[0]);
}

__global__ __launch_bounds__(256) void k_msel1(const float* __restrict__ pred,
    const float* __restrict__ coord, float* __restrict__ accSq, float* __restrict__ accL1) {
  int idx = blockIdx.x * 256 + threadIdx.x;
  float sq = 0.f, l1 = 0.f;
  if (idx < MT * 3) {
    int i = idx / 3, c = idx - i * 3;
    float d = pred[(size_t)i * 16 + c] - coord[idx];
    sq = d * d;
    float a = fabsf(d);
    l1 = (a < 1.f) ? 0.5f * d * d : (a - 0.5f);
  }
  __shared__ float s1[256], s2[256];
  s1[threadIdx.x] = sq; s2[threadIdx.x] = l1; __syncthreads();
  for (int off = 128; off > 0; off >>= 1) {
    if (threadIdx.x < off) { s1[threadIdx.x] += s1[threadIdx.x + off]; s2[threadIdx.x] += s2[threadIdx.x + off]; }
    __syncthreads();
  }
  if (threadIdx.x == 0) { atomicAdd(accSq, s1[0]); atomicAdd(accL1, s2[0]); }
}

__global__ void k_final(const float* __restrict__ acc, const float* __restrict__ maxB,
                        float* __restrict__ out) {
  if (threadIdx.x != 0 || blockIdx.x != 0) return;
  float chamfer = (acc[0] + acc[1]) * (1.f / MT);
  float h = 0.f;
  for (int b = 0; b < 8; ++b) h += fmaxf(maxB[b], maxB[8 + b]);
  h *= 0.125f;
  float mse = acc[3] * (1.f / (MT * 3));
  float sm = sqrtf(mse);
  float l1 = acc[4] * (1.f / (MT * 3));
  float rep = acc[2] * (1.f / (MT * 5));
  out[0] = 10.f * sm; out[1] = chamfer; out[2] = rep; out[3] = sm; out[4] = h; out[5] = l1;
}

// ---------------- host orchestration ----------------

static inline dim3 grid1(long n) { return dim3((unsigned)((n + 255) / 256)); }

extern "C" void kernel_launch(void* const* d_in, const int* in_sizes, int n_in,
                              void* d_out, int out_size, void* d_ws, size_t ws_size,
                              hipStream_t stream) {
  (void)out_size; (void)ws_size;
  const float *pc, *coord;
  const float *pW[3], *pb[3], *pg[3], *pbt[3];
  const float *cW[4], *cb[4], *cg[3], *cbt[3];
  if (n_in >= 28 && in_sizes[2] == 192) {
    // insertion order: pc, coord, {pW0,pb0,pg0,pbeta0,...,cW3,cb3}
    pc = (const float*)d_in[0]; coord = (const float*)d_in[1];
    int k = 2;
    for (int i = 0; i < 3; ++i) { pW[i]=(const float*)d_in[k++]; pb[i]=(const float*)d_in[k++];
                                  pg[i]=(const float*)d_in[k++]; pbt[i]=(const float*)d_in[k++]; }
    for (int i = 0; i < 4; ++i) { cW[i]=(const float*)d_in[k++]; cb[i]=(const float*)d_in[k++];
                                  if (i < 3) { cg[i]=(const float*)d_in[k++]; cbt[i]=(const float*)d_in[k++]; } }
  } else {
    // jax pytree sorted-leaf order: coord, cW0..3, cb0..3, cbeta0..2, cg0..2, pW0..2, pb0..2, pbeta0..2, pg0..2, pc
    coord = (const float*)d_in[0];
    for (int i = 0; i < 4; ++i) cW[i] = (const float*)d_in[1 + i];
    for (int i = 0; i < 4; ++i) cb[i] = (const float*)d_in[5 + i];
    for (int i = 0; i < 3; ++i) cbt[i] = (const float*)d_in[9 + i];
    for (int i = 0; i < 3; ++i) cg[i]  = (const float*)d_in[12 + i];
    for (int i = 0; i < 3; ++i) pW[i]  = (const float*)d_in[15 + i];
    for (int i = 0; i < 3; ++i) pb[i]  = (const float*)d_in[18 + i];
    for (int i = 0; i < 3; ++i) pbt[i] = (const float*)d_in[21 + i];
    for (int i = 0; i < 3; ++i) pg[i]  = (const float*)d_in[24 + i];
    pc = (const float*)d_in[27];
  }

  char* w = (char*)d_ws;
  auto alloc = [&](size_t bytes) -> void* { void* p = (void*)w; w += (bytes + 255) & ~(size_t)255; return p; };

  float* acc   = (float*)alloc(64 * sizeof(float));   // [0]=sum12 [1]=sum21 [2]=rep [3]=sq [4]=l1
  float* maxB  = acc + 16;                            // 16 per-batch maxima
  float* scale = (float*)alloc(1024 * sizeof(float));
  float* shift = (float*)alloc(1024 * sizeof(float));
  float* partB = (float*)alloc(32 * 1024 * sizeof(float));
  float* gbuf  = (float*)alloc(8 * 1024 * sizeof(float));
  float* cb3p  = (float*)alloc(16 * sizeof(float));
  float* m12   = (float*)alloc(MT * sizeof(float));
  float* m21   = (float*)alloc(MT * sizeof(float));
  float* predP = (float*)alloc((size_t)MT * 16 * sizeof(float));
  float* Ybuf  = (float*)alloc((size_t)MT * 1024 * sizeof(float));
  _Float16* X0h  = (_Float16*)alloc((size_t)MT * 32 * 2);
  _Float16* A1h  = (_Float16*)alloc((size_t)MT * 64 * 2);
  _Float16* A2h  = (_Float16*)alloc((size_t)MT * 128 * 2);
  _Float16* Xch  = (_Float16*)alloc((size_t)MT * 1088 * 2);
  _Float16* A4h  = (_Float16*)alloc((size_t)MT * 512 * 2);
  _Float16* A5h  = (_Float16*)alloc((size_t)MT * 256 * 2);
  _Float16* A6h  = (_Float16*)alloc((size_t)MT * 128 * 2);
  _Float16* pW0h = (_Float16*)alloc(64 * 32 * 2);
  _Float16* pW1h = (_Float16*)alloc(128 * 64 * 2);
  _Float16* pW2h = (_Float16*)alloc((size_t)1024 * 128 * 2);
  _Float16* cW0h = (_Float16*)alloc((size_t)512 * 1088 * 2);
  _Float16* cW1h = (_Float16*)alloc((size_t)256 * 512 * 2);
  _Float16* cW2h = (_Float16*)alloc((size_t)128 * 256 * 2);
  _Float16* cW3h = (_Float16*)alloc(16 * 128 * 2);

  k_zerof<<<1, 64, 0, stream>>>(acc, 64);
  k_pad_w<<<grid1(64 * 32),    256, 0, stream>>>(pW[0], pW0h, 64, 3, 64, 32);
  k_pad_w<<<grid1(128 * 64),   256, 0, stream>>>(pW[1], pW1h, 128, 64, 128, 64);
  k_pad_w<<<grid1(1024 * 128), 256, 0, stream>>>(pW[2], pW2h, 1024, 128, 1024, 128);
  k_pad_w<<<grid1((long)512 * 1088), 256, 0, stream>>>(cW[0], cW0h, 512, 1088, 512, 1088);
  k_pad_w<<<grid1(256 * 512),  256, 0, stream>>>(cW[1], cW1h, 256, 512, 256, 512);
  k_pad_w<<<grid1(128 * 256),  256, 0, stream>>>(cW[2], cW2h, 128, 256, 128, 256);
  k_pad_w<<<grid1(16 * 128),   256, 0, stream>>>(cW[3], cW3h, 3, 128, 16, 128);
  k_pad_b<<<1, 64, 0, stream>>>(cb[3], cb3p, 3, 16);
  k_x0<<<grid1((long)MT * 32), 256, 0, stream>>>(pc, X0h);

  auto gemm = [&](const _Float16* A, const _Float16* Wh, const float* bias, float* Y, int Npad, int K) {
    if ((Npad & 63) == 0) {
      int tiles = (MT / 32) * (Npad / 64);
      k_gemm<4><<<dim3((tiles + 7) / 8), 256, 0, stream>>>(A, Wh, bias, Y, MT, Npad, K);
    } else {
      int tiles = (MT / 32) * (Npad / 16);
      k_gemm<1><<<dim3((tiles + 7) / 8), 256, 0, stream>>>(A, Wh, bias, Y, MT, Npad, K);
    }
  };
  auto bn = [&](const float* g_, const float* b_, int C) {
    k_bnpart<<<dim3(C / 16, 16), 256, 0, stream>>>(Ybuf, partB, C);
    k_bnfin<<<grid1(C), 256, 0, stream>>>(partB, g_, b_, scale, shift, C);
  };

  // point MLP: 3->64->128->1024, BN+ReLU each
  gemm(X0h, pW0h, pb[0], Ybuf, 64, 32);
  bn(pg[0], pbt[0], 64);
  k_bnrelu<<<grid1((long)MT * 64), 256, 0, stream>>>(Ybuf, scale, shift, A1h, 64, Xch, 1088, 0, 64);

  gemm(A1h, pW1h, pb[1], Ybuf, 128, 64);
  bn(pg[1], pbt[1], 128);
  k_bnrelu<<<grid1((long)MT * 128), 256, 0, stream>>>(Ybuf, scale, shift, A2h, 128, (_Float16*)nullptr, 0, 0, 128);

  gemm(A2h, pW2h, pb[2], Ybuf, 1024, 128);
  bn(pg[2], pbt[2], 1024);
  k_maxpool<<<dim3(64, 8), 256, 0, stream>>>(Ybuf, scale, shift, gbuf);
  k_bcast<<<grid1((long)MT * 1024), 256, 0, stream>>>(gbuf, Xch);

  // coord MLP: 1088->512->256->128->3 (last: plain conv, Cout padded to 16)
  gemm(Xch, cW0h, cb[0], Ybuf, 512, 1088);
  bn(cg[0], cbt[0], 512);
  k_bnrelu<<<grid1((long)MT * 512), 256, 0, stream>>>(Ybuf, scale, shift, A4h, 512, (_Float16*)nullptr, 0, 0, 512);

  gemm(A4h, cW1h, cb[1], Ybuf, 256, 512);
  bn(cg[1], cbt[1], 256);
  k_bnrelu<<<grid1((long)MT * 256), 256, 0, stream>>>(Ybuf, scale, shift, A5h, 256, (_Float16*)nullptr, 0, 0, 256);

  gemm(A5h, cW2h, cb[2], Ybuf, 128, 256);
  bn(cg[2], cbt[2], 128);
  k_bnrelu<<<grid1((long)MT * 128), 256, 0, stream>>>(Ybuf, scale, shift, A6h, 128, (_Float16*)nullptr, 0, 0, 128);

  gemm(A6h, cW3h, cb3p, predP, 16, 128);   // pred in cols 0..2 of stride-16 rows

  // losses
  k_minrows<<<MT, 256, 0, stream>>>(predP, 16, coord, 3, m12);
  k_minrows<<<MT, 256, 0, stream>>>(coord, 3, predP, 16, m21);
  k_summax<<<16, 256, 0, stream>>>(m12, m21, acc, maxB);
  k_rep<<<MT / 256, 256, 0, stream>>>(predP, &acc[2]);
  k_msel1<<<grid1((long)MT * 3), 256, 0, stream>>>(predP, coord, &acc[3], &acc[4]);
  k_final<<<1, 32, 0, stream>>>(acc, maxB, (float*)d_out);
}
